// ItemAutoencoder_5892695130646
// MI455X (gfx1250) — matrix operand order repaired and checked by reading, not compiled, not run on hardware
//
#include <hip/hip_runtime.h>
#include <hip/hip_bf16.h>

typedef __attribute__((ext_vector_type(16))) _Float16 v16h;
typedef __attribute__((ext_vector_type(8)))  _Float16 v8h;
typedef __attribute__((ext_vector_type(8)))  float    v8f;
typedef int v4i_vs __attribute__((vector_size(16)));

// ---------------------------------------------------------------------------
// Fast transcendentals: CDNA5 has a hardware V_TANH_F32 (TRANS32 class).
// sigmoid(x) = 0.5 + 0.5 * tanh(x/2)
// ---------------------------------------------------------------------------
__device__ __forceinline__ float fast_tanh(float x) {
#if __has_builtin(__builtin_amdgcn_tanhf)
  return __builtin_amdgcn_tanhf(x);
#elif __has_builtin(__builtin_amdgcn_tanh_f32)
  return __builtin_amdgcn_tanh_f32(x);
#else
  float r;
  asm volatile("v_tanh_f32 %0, %1\n\ts_delay_alu instid0(TRANS32_DEP_1)"
               : "=v"(r)
               : "v"(x));
  return r;
#endif
}
__device__ __forceinline__ float fast_sig(float x) {
  return fmaf(0.5f, fast_tanh(0.5f * x), 0.5f);
}

// ---------------------------------------------------------------------------
// Async global->LDS staging (ASYNCcnt path) so x-tile staging does not sit in
// the in-order LOADcnt stream that feeds the WMMA weight pipeline.
// Builtin signature (from clang diagnostic): (global int4*, lds int4*, Ii, Ii)
// ---------------------------------------------------------------------------
#if __has_builtin(__builtin_amdgcn_global_load_async_to_lds_b128) && \
    __has_builtin(__builtin_amdgcn_s_wait_asynccnt)
#define HAVE_ASYNC_LDS 1
__device__ __forceinline__ void async_cp16(const _Float16* g, _Float16* l) {
  __builtin_amdgcn_global_load_async_to_lds_b128(
      (__attribute__((address_space(1))) v4i_vs*)(void*)g,
      (__attribute__((address_space(3))) v4i_vs*)(void*)l, 0, 0);
}
__device__ __forceinline__ void async_wait0() { __builtin_amdgcn_s_wait_asynccnt(0); }
#else
#define HAVE_ASYNC_LDS 0
__device__ __forceinline__ void async_wait0() {}
#endif

// ---------------------------------------------------------------------------
// WMMA fragment loaders (CDNA5 v_wmma_f32_16x16x32_f16 layouts, wave32)
// A 16x32 f16: lanes 0-15 row M=lane, elems 0..7 = K 0..7, elems 8..15 = K 16..23
//              lanes 16-31 row M=lane-16, elems 0..7 = K 8..15, elems 8..15 = K 24..31
// B 32x16 f16: lanes 0-15 col N=lane, elems 0..7 = K 0..7, 8..15 = K 16..23; lanes 16-31 col N=lane-16, K 8..15, 24..31
// C/D 16x16 f32: vgpr e -> M = e (lanes 0-15) / e+8 (lanes 16-31), N = lane&15
// ---------------------------------------------------------------------------
__device__ __forceinline__ v16h load_a_frag(const _Float16* tile, int ld, int lane) {
  int r    = lane & 15;
  int koff = (lane < 16) ? 0 : 8;
  const _Float16* p = tile + r * ld + koff;
  v8h lo = *(const v8h*)(p);
  v8h hi = *(const v8h*)(p + 16);
  return __builtin_shufflevector(lo, hi, 0, 1, 2, 3, 4, 5, 6, 7, 8, 9, 10, 11, 12, 13, 14, 15);
}

// W is (N, K) row-major f16 (column n of B = row n of W)
__device__ __forceinline__ v16h load_b_frag(const _Float16* W, int ldk, int n0, int k0, int lane) {
  int n  = lane & 15;
  int ko = (lane < 16) ? 0 : 8;
  const _Float16* p = W + (size_t)(n0 + n) * ldk + (k0 + ko);
  v8h lo = *(const v8h*)(p);
  v8h hi = *(const v8h*)(p + 16);
  return __builtin_shufflevector(lo, hi, 0, 1, 2, 3, 4, 5, 6, 7, 8, 9, 10, 11, 12, 13, 14, 15);
}

__device__ __forceinline__ v8f wmma_f16(v16h a, v16h b, v8f c) {
  return __builtin_amdgcn_wmma_f32_16x16x32_f16(false, a, false, b, (short)0, c, false, false);
}

// ---------------------------------------------------------------------------
// Conversion / bias helpers
// ---------------------------------------------------------------------------
__global__ void cvt_f32_f16(const float* __restrict__ src, _Float16* __restrict__ dst, int n) {
  int i = blockIdx.x * 256 + threadIdx.x;
  if (i < n) dst[i] = (_Float16)src[i];
}

// Permute LSTM weight rows so that each 64-row block holds i,f,g,o for 16 units:
// src row r = g*HID + j  ->  dst row = (j/16)*64 + g*16 + (j%16)
__global__ void cvt_permute_gate(const float* __restrict__ src, _Float16* __restrict__ dst,
                                 int HID, int K) {
  int idx = blockIdx.x * 256 + threadIdx.x;
  if (idx >= 4 * HID * K) return;
  int r = idx / K, k = idx - r * K;
  int g = r / HID, j = r - g * HID;
  int nr = (j >> 4) * 64 + g * 16 + (j & 15);
  dst[(size_t)nr * K + k] = (_Float16)src[idx];
}

// Combined (bih + bhh) bias with the same gate-row permutation.
__global__ void bias_permute(const float* __restrict__ a, const float* __restrict__ b,
                             float* __restrict__ dst, int HID) {
  int idx = blockIdx.x * 256 + threadIdx.x;
  if (idx >= 4 * HID) return;
  int g = idx / HID, j = idx - g * HID;
  int nr = (j >> 4) * 64 + g * 16 + (j & 15);
  dst[nr] = a[idx] + b[idx];
}

// ---------------------------------------------------------------------------
// Generic WMMA GEMM:  out[r][n] = act( sum_k A[r][k] * W[n][k] + bias[n] )
// A: (R, K) f16 row-major, W: (N, K) f16 row-major, K % 32 == 0.
// 4 waves / block, one 16x16 tile per wave. grid = (R/64, N/16).
// ---------------------------------------------------------------------------
template <int RELU, typename OUT>
__global__ __launch_bounds__(128) void gemm16(const _Float16* __restrict__ A,
                                              const _Float16* __restrict__ W,
                                              const float* __restrict__ bias,
                                              OUT* __restrict__ out, int K, int N) {
  int wave  = threadIdx.x >> 5;
  int lane  = threadIdx.x & 31;
  int tileR = blockIdx.x * 4 + wave;
  int n0    = blockIdx.y * 16;

  const _Float16* Abase = A + (size_t)tileR * 16 * K;
  v8f acc = {};
  for (int k0 = 0; k0 < K; k0 += 32) {
    v16h a = load_a_frag(Abase + k0, K, lane);
    v16h b = load_b_frag(W, K, n0, k0, lane);
    acc = wmma_f16(a, b, acc);
  }
  float bv = bias[n0 + (lane & 15)];
  int mb = (lane < 16) ? 0 : 8;
  int n  = n0 + (lane & 15);
#pragma unroll
  for (int e = 0; e < 8; ++e) {
    float v = acc[e] + bv;
    if (RELU) v = fmaxf(v, 0.0f);
    out[((size_t)tileR * 16 + mb + e) * N + n] = (OUT)v;
  }
}

// ---------------------------------------------------------------------------
// Persistent fused LSTM layer (gate-permuted weights).
// One workgroup = 16 batch rows for all T steps.  NWAVES waves; each wave owns
// GPW = HID/16/NWAVES groups of 16 hidden units (4 gate tiles each), so the
// i/f/g/o for a unit land in the same (lane, elem) of 4 accumulators and the
// cell state c stays in VGPRs for the whole sequence.
// Double-buffered s_x / s_h in LDS -> one barrier per step; x(t+1) is staged
// via async global->LDS copies (ASYNCcnt) overlapping the step's WMMAs.
// BCAST=1: input constant over t (decoder z); its gate contribution + bias is
// hoisted into registers. LAST_ONLY=1: only h(T-1) is written out (encoder top).
// Weights preloaded into VGPRs when the per-wave footprint is <= 128 regs.
// ---------------------------------------------------------------------------
template <int IN, int HID, int NWAVES, int BCAST, int LAST_ONLY>
__global__ __launch_bounds__(NWAVES * 32) void lstm16(
    const _Float16* __restrict__ xin, int xRowStride, int xTStride,
    const _Float16* __restrict__ Wi, const _Float16* __restrict__ Wh,
    const float* __restrict__ bias, _Float16* __restrict__ hout, int T) {
  constexpr int NTHREADS = NWAVES * 32;
  constexpr int GPW = HID / 16 / NWAVES;  // 16-unit groups per wave
  constexpr int KI = IN / 32, KH = HID / 32;
  constexpr int CHX = IN / 8;  // 16-byte chunks per x row
  constexpr bool PRE_WI = (!BCAST) && (KI * GPW * 4 <= 16);  // <=128 VGPRs
  constexpr bool PRE_WH = (KH * GPW * 4 <= 16);

  extern __shared__ char smem[];
  _Float16* s_x0 = (_Float16*)smem;       // 16*IN
  _Float16* s_x1 = s_x0 + 16 * IN;        // 16*IN
  _Float16* s_h0 = s_x1 + 16 * IN;        // 16*HID
  _Float16* s_h1 = s_h0 + 16 * HID;       // 16*HID

  int b0   = blockIdx.x * 16;
  int tid  = threadIdx.x;
  int wave = tid >> 5;
  int lane = tid & 31;
  int lmb  = (lane < 16) ? 0 : 8;  // row base of this half-wave in C layout
  int lno  = lane & 15;            // column within tile

  // zero h(0); stage x(0) (vectorized 16B copies)
  for (int i = tid; i < 16 * HID; i += NTHREADS) s_h0[i] = (_Float16)0.0f;
  for (int i = tid; i < 16 * CHX; i += NTHREADS) {
    int r = i / CHX, c = i - r * CHX;
    *(v8h*)(s_x0 + r * IN + c * 8) =
        *(const v8h*)(xin + (size_t)(b0 + r) * xRowStride + c * 8);
  }
  __syncthreads();

  // optional register-resident weights
  v16h wiF[PRE_WI ? KI * GPW * 4 : 1];
  v16h whF[PRE_WH ? KH * GPW * 4 : 1];
  if constexpr (PRE_WI) {
#pragma unroll
    for (int tile = 0; tile < GPW * 4; ++tile)
#pragma unroll
      for (int k = 0; k < KI; ++k)
        wiF[tile * KI + k] = load_b_frag(Wi, IN, (wave * GPW * 4 + tile) * 16, k * 32, lane);
  }
  if constexpr (PRE_WH) {
#pragma unroll
    for (int tile = 0; tile < GPW * 4; ++tile)
#pragma unroll
      for (int k = 0; k < KH; ++k)
        whF[tile * KH + k] = load_b_frag(Wh, HID, (wave * GPW * 4 + tile) * 16, k * 32, lane);
  }

  // bias preloaded to registers
  float biasReg[GPW * 4];
#pragma unroll
  for (int tile = 0; tile < GPW * 4; ++tile)
    biasReg[tile] = bias[(wave * GPW * 4 + tile) * 16 + lno];

  // BCAST: hoist constant-input gate contribution (+bias) out of the T-loop
  v8f base[BCAST ? GPW * 4 : 1];
  if constexpr (BCAST) {
#pragma unroll
    for (int tile = 0; tile < GPW * 4; ++tile) {
      int n0 = (wave * GPW * 4 + tile) * 16;
      v8f a;
#pragma unroll
      for (int e = 0; e < 8; ++e) a[e] = biasReg[tile];
#pragma unroll
      for (int k = 0; k < KI; ++k)
        a = wmma_f16(load_a_frag(s_x0 + k * 32, IN, lane),
                     load_b_frag(Wi, IN, n0, k * 32, lane), a);
      base[tile] = a;
    }
  }

  float creg[GPW][8] = {};  // cell state, register-resident

  for (int t = 0; t < T; ++t) {
    const _Float16* s_x  = (t & 1) ? s_x1 : s_x0;
    const _Float16* s_hr = (t & 1) ? s_h1 : s_h0;
    _Float16* s_hw = (t & 1) ? s_h0 : s_h1;
    _Float16* s_xw = (t & 1) ? s_x0 : s_x1;

    // stage next x tile into the other buffer, overlapping this step's math
    if constexpr (!BCAST) {
      if (t + 1 < T) {
        for (int i = tid; i < 16 * CHX; i += NTHREADS) {
          int r = i / CHX, c = i - r * CHX;
          const _Float16* src =
              xin + (size_t)(b0 + r) * xRowStride + (size_t)(t + 1) * xTStride + c * 8;
          _Float16* dst = s_xw + r * IN + c * 8;
#if HAVE_ASYNC_LDS
          async_cp16(src, dst);
#else
          *(v8h*)dst = *(const v8h*)src;
#endif
        }
      }
    }

#pragma unroll
    for (int u = 0; u < GPW; ++u) {
      int ug = wave * GPW + u;
      v8f acc[4];
#pragma unroll
      for (int g = 0; g < 4; ++g) {
        int n0 = (ug * 4 + g) * 16;
        v8f a;
        if constexpr (BCAST) {
          a = base[u * 4 + g];
        } else {
          float bv = biasReg[u * 4 + g];
#pragma unroll
          for (int e = 0; e < 8; ++e) a[e] = bv;
#pragma unroll
          for (int k = 0; k < KI; ++k) {
            v16h av = load_a_frag(s_x + k * 32, IN, lane);
            v16h bf;
            if constexpr (PRE_WI) bf = wiF[(u * 4 + g) * KI + k];
            else bf = load_b_frag(Wi, IN, n0, k * 32, lane);
            a = wmma_f16(av, bf, a);
          }
        }
#pragma unroll
        for (int k = 0; k < KH; ++k) {
          v16h av = load_a_frag(s_hr + k * 32, HID, lane);
          v16h bf;
          if constexpr (PRE_WH) bf = whF[(u * 4 + g) * KH + k];
          else bf = load_b_frag(Wh, HID, n0, k * 32, lane);
          a = wmma_f16(av, bf, a);
        }
        acc[g] = a;
      }
      // in-register cell update: acc[0]=i, acc[1]=f, acc[2]=g, acc[3]=o
      int j = ug * 16 + lno;
#pragma unroll
      for (int e = 0; e < 8; ++e) {
        float cv = fast_sig(acc[1][e]) * creg[u][e] + fast_sig(acc[0][e]) * fast_tanh(acc[2][e]);
        float hv = fast_sig(acc[3][e]) * fast_tanh(cv);
        creg[u][e] = cv;
        _Float16 h16 = (_Float16)hv;
        s_hw[(lmb + e) * HID + j] = h16;
        if constexpr (LAST_ONLY) {
          if (t == T - 1) hout[(size_t)(b0 + lmb + e) * HID + j] = h16;
        } else {
          hout[((size_t)(b0 + lmb + e) * T + t) * HID + j] = h16;
        }
      }
    }
#if HAVE_ASYNC_LDS
    if constexpr (!BCAST) async_wait0();  // x(t+1) landed in LDS
#endif
    __syncthreads();  // h(t+1), x(t+1) visible; safe to reuse buffers
  }
}

static constexpr size_t lstm_smem(int IN, int HID) {
  return (size_t)64 * IN + (size_t)64 * HID;  // two f16 x-bufs + two f16 h-bufs
}

// ---------------------------------------------------------------------------
// Host launcher
// ---------------------------------------------------------------------------
extern "C" void kernel_launch(void* const* d_in, const int* in_sizes, int n_in,
                              void* d_out, int out_size, void* d_ws, size_t ws_size,
                              hipStream_t stream) {
  const float* x     = (const float*)d_in[0];
  const float* in_W  = (const float*)d_in[1];
  const float* in_b  = (const float*)d_in[2];
  const float* eW0i  = (const float*)d_in[3];
  const float* eW0h  = (const float*)d_in[4];
  const float* eb0i  = (const float*)d_in[5];
  const float* eb0h  = (const float*)d_in[6];
  const float* eW1i  = (const float*)d_in[7];
  const float* eW1h  = (const float*)d_in[8];
  const float* eb1i  = (const float*)d_in[9];
  const float* eb1h  = (const float*)d_in[10];
  const float* dW0i  = (const float*)d_in[11];
  const float* dW0h  = (const float*)d_in[12];
  const float* db0i  = (const float*)d_in[13];
  const float* db0h  = (const float*)d_in[14];
  const float* dW1i  = (const float*)d_in[15];
  const float* dW1h  = (const float*)d_in[16];
  const float* db1i  = (const float*)d_in[17];
  const float* db1h  = (const float*)d_in[18];
  const float* out_W = (const float*)d_in[19];
  const float* out_b = (const float*)d_in[20];
  float* out = (float*)d_out;

  constexpr int B = 1024, T = 100, D = 256, H = 256, M = 64;
  constexpr size_t BT = (size_t)B * T;

  char* ws = (char*)d_ws;
  size_t o = 0;
  auto carve = [&](size_t bytes) -> char* {
    char* p = ws + o;
    o = (o + bytes + 255) & ~(size_t)255;
    return p;
  };
  // big sequence buffers (reused across stages)
  _Float16* bufA = (_Float16*)carve(BT * D * 2);   // x16 -> enc0 out -> dec1 out
  _Float16* bufZ = (_Float16*)carve((size_t)B * H * 2);  // z = enc1 h(T-1)
  _Float16* bufC = (_Float16*)carve(BT * M * 2);   // in-proj out -> dec0 out
  // f16 weights (LSTM ones stored gate-permuted)
  _Float16* w_in  = (_Float16*)carve((size_t)M * D * 2);
  _Float16* w_e0i = (_Float16*)carve((size_t)4 * H * M * 2);
  _Float16* w_e0h = (_Float16*)carve((size_t)4 * H * H * 2);
  _Float16* w_e1i = (_Float16*)carve((size_t)4 * H * H * 2);
  _Float16* w_e1h = (_Float16*)carve((size_t)4 * H * H * 2);
  _Float16* w_d0i = (_Float16*)carve((size_t)4 * M * H * 2);
  _Float16* w_d0h = (_Float16*)carve((size_t)4 * M * M * 2);
  _Float16* w_d1i = (_Float16*)carve((size_t)4 * M * M * 2);
  _Float16* w_d1h = (_Float16*)carve((size_t)4 * M * M * 2);
  _Float16* w_out = (_Float16*)carve((size_t)D * M * 2);
  // combined, gate-permuted biases
  float* b_e0 = (float*)carve(4 * H * 4);
  float* b_e1 = (float*)carve(4 * H * 4);
  float* b_d0 = (float*)carve(4 * M * 4);
  float* b_d1 = (float*)carve(4 * M * 4);

  auto cvt = [&](const float* s, _Float16* d, size_t n) {
    cvt_f32_f16<<<dim3((unsigned)((n + 255) / 256)), dim3(256), 0, stream>>>(s, d, (int)n);
  };
  auto cvtp = [&](const float* s, _Float16* d, int HID_, int K_) {
    size_t n = (size_t)4 * HID_ * K_;
    cvt_permute_gate<<<dim3((unsigned)((n + 255) / 256)), dim3(256), 0, stream>>>(s, d, HID_, K_);
  };
  cvt(x, bufA, BT * D);
  cvt(in_W, w_in, (size_t)M * D);
  cvt(out_W, w_out, (size_t)D * M);
  cvtp(eW0i, w_e0i, H, M);
  cvtp(eW0h, w_e0h, H, H);
  cvtp(eW1i, w_e1i, H, H);
  cvtp(eW1h, w_e1h, H, H);
  cvtp(dW0i, w_d0i, M, H);
  cvtp(dW0h, w_d0h, M, M);
  cvtp(dW1i, w_d1i, M, M);
  cvtp(dW1h, w_d1h, M, M);

  bias_permute<<<dim3(4), dim3(256), 0, stream>>>(eb0i, eb0h, b_e0, H);
  bias_permute<<<dim3(4), dim3(256), 0, stream>>>(eb1i, eb1h, b_e1, H);
  bias_permute<<<dim3(1), dim3(256), 0, stream>>>(db0i, db0h, b_d0, M);
  bias_permute<<<dim3(1), dim3(256), 0, stream>>>(db1i, db1h, b_d1, M);

  // input projection + ReLU: (BT, D) @ (M, D)^T -> (BT, M) f16
  gemm16<1, _Float16><<<dim3((unsigned)(BT / 64), M / 16), dim3(128), 0, stream>>>(
      bufA, w_in, in_b, bufC, D, M);

  // encoder LSTM 0: IN=64, HID=256, 8 waves (Wi register-resident)
  lstm16<64, 256, 8, 0, 0><<<dim3(B / 16), dim3(256), lstm_smem(64, 256), stream>>>(
      bufC, T * M, M, w_e0i, w_e0h, b_e0, bufA, T);

  // encoder LSTM 1: IN=256, HID=256, 8 waves; only h(T-1) -> z buffer
  lstm16<256, 256, 8, 0, 1><<<dim3(B / 16), dim3(256), lstm_smem(256, 256), stream>>>(
      bufA, T * H, H, w_e1i, w_e1h, b_e1, bufZ, T);

  // decoder LSTM 0: broadcast z; z-gates hoisted once, Wh register-resident
  lstm16<256, 64, 4, 1, 0><<<dim3(B / 16), dim3(128), lstm_smem(256, 64), stream>>>(
      bufZ, H, 0, w_d0i, w_d0h, b_d0, bufC, T);

  // decoder LSTM 1: IN=64, HID=64, 4 waves (Wi and Wh register-resident)
  lstm16<64, 64, 4, 0, 0><<<dim3(B / 16), dim3(128), lstm_smem(64, 64), stream>>>(
      bufC, T * M, M, w_d1i, w_d1h, b_d1, bufA, T);

  // output projection: (BT, M) @ (D, M)^T + out_b -> d_out f32
  gemm16<0, float><<<dim3((unsigned)(BT / 64), D / 16), dim3(128), 0, stream>>>(
      bufA, w_out, out_b, out, M, D);
}
